// SchNetLayer_13340168421781
// MI455X (gfx1250) — matrix operand order repaired and checked by reading, not compiled, not run on hardware
//
#include <hip/hip_runtime.h>

// ---------------- problem constants ----------------
constexpr int NB     = 2048;          // batch
constexpr int NNUC   = 4;
constexpr int NELEC  = 16;
constexpr int NPART  = NNUC + NELEC;  // 20
constexpr int NNODES = NB * NPART;    // 40960
constexpr int NE     = 307200;        // edges
constexpr int DD     = 64;            // dist feature dim
constexpr int KF     = 128;           // message dim
constexpr int EMBD   = 256;           // embedding dim

// ---------------- WMMA types ----------------
typedef __attribute__((ext_vector_type(16))) __bf16 v16bf;
typedef __attribute__((ext_vector_type(8)))  float  v8f;

union ABf { v16bf v; unsigned short u[16]; };

__device__ __forceinline__ unsigned short f2bf(float f) {
    union { float f; unsigned u; } x; x.f = f;
    unsigned u = x.u + 0x7FFFu + ((x.u >> 16) & 1u);   // round-to-nearest-even
    return (unsigned short)(u >> 16);
}

__device__ __forceinline__ float ssp(float x) {
    // softplus(x) - log(2), numerically stable
    float sp = fmaxf(x, 0.0f) + log1pf(__expf(-fabsf(x)));
    return sp - 0.6931471805599453f;
}

__device__ __forceinline__ v8f wmma_bf16(v16bf a, v16bf b, v8f c) {
    return __builtin_amdgcn_wmma_f32_16x16x32_bf16(
        /*neg_a=*/false, a, /*neg_b=*/false, b,
        /*c_mod=*/(short)0, c, /*reuse_a=*/false, /*reuse_b=*/false);
}

// A fragment (bf16 16x32) from LDS, ISA 7.12.2 layout:
//   m = lane%16, K-offset = 8*(lane>=16); VGPR halves at K+0..7 / K+16..23(+kb)
__device__ __forceinline__ v16bf load_a_frag(const unsigned short* a_lds, int lda,
                                             int k0, int lane) {
    int m  = lane & 15;
    int kb = (lane >> 4) << 3;              // 0 or 8
    ABf f;
#pragma unroll
    for (int h = 0; h < 2; ++h)
#pragma unroll
        for (int j = 0; j < 8; ++j)
            f.u[h * 8 + j] = a_lds[m * lda + k0 + h * 16 + kb + j];
    return f.v;
}

// B fragment (bf16 32x16) from transposed-bf16 weights Bt[n*ldk + k]:
//   n = n0 + lane%16, K-offset = 8*(lane>=16); halves at K+0..7 / K+16..23
__device__ __forceinline__ v16bf load_b_frag(const unsigned short* bt, int ldk,
                                             int n0, int k0, int lane) {
    const unsigned short* p = bt + (size_t)(n0 + (lane & 15)) * ldk + k0 + ((lane >> 4) << 3);
    ABf f;
#pragma unroll
    for (int e = 0; e < 16; ++e) f.u[e] = p[e + (e & 8)];
    return f.v;
}

// ---------------- kernel: transpose-convert f32 weights -> bf16 ----------------
__global__ void cvt_transpose(const float* __restrict__ src, unsigned short* __restrict__ dst,
                              int rowsK, int colsN) {
    int i = blockIdx.x * blockDim.x + threadIdx.x;
    if (i < rowsK * colsN) {
        int r = i / colsN, c = i % colsN;
        dst[c * rowsK + r] = f2bf(src[i]);
    }
}

// ---------------- kernel: copy nuc rows into hx ----------------
__global__ void copy_nuc(const float* __restrict__ nuc, float* __restrict__ hx) {
    int i = blockIdx.x * blockDim.x + threadIdx.x;    // over NB*4*KF
    if (i < NB * NNUC * KF) {
        int b = i / (NNUC * KF), rem = i % (NNUC * KF);
        hx[(size_t)b * NPART * KF + rem] = nuc[i];
    }
}

// ---------------- kernel: hx electron rows = elec @ h_W + h_b ----------------
// one batch (16 rows x 256 K) per block; 8 waves, each a 16-col tile of N=128... 8*16=128
__global__ void hx_gemm(const float* __restrict__ elec, const unsigned short* __restrict__ hWt,
                        const float* __restrict__ hb, float* __restrict__ hx) {
    constexpr int LDA = EMBD + 8;
    __shared__ unsigned short a_lds[16 * LDA];
    int b = blockIdx.x, tid = threadIdx.x, lane = tid & 31, w = tid >> 5;

    const float* src = elec + (size_t)b * NELEC * EMBD;
    for (int i = tid; i < 16 * EMBD; i += 256) {
        int r = i >> 8, c = i & 255;
        a_lds[r * LDA + c] = f2bf(src[i]);
    }
    __syncthreads();

    int n0 = w * 16, n = n0 + (lane & 15);
    v8f c; float bias = hb[n];
#pragma unroll
    for (int r = 0; r < 8; ++r) c[r] = bias;

    for (int k0 = 0; k0 < EMBD; k0 += 32) {
        v16bf a  = load_a_frag(a_lds, LDA, k0, lane);
        v16bf bb = load_b_frag(hWt, EMBD, n0, k0, lane);
        c = wmma_bf16(a, bb, c);
    }
    int mh = (lane >> 4) * 8;
    float* dst = hx + ((size_t)b * NPART + NNUC) * KF;
#pragma unroll
    for (int r = 0; r < 8; ++r) dst[(size_t)(mh + r) * KF + n] = c[r];
}

// ---------------- kernel: bucket edges by type (3 -> 0 same, 4 -> 1 anti, 1 -> 2 nuc) ----
__global__ void compact_edges(const int* __restrict__ e_type, int* __restrict__ counts,
                              int* __restrict__ elist) {
    int i = blockIdx.x * blockDim.x + threadIdx.x;
    if (i < NE) {
        int et = e_type[i];
        int t  = (et == 3) ? 0 : ((et == 4) ? 1 : 2);
        int p  = atomicAdd(&counts[t], 1);
        elist[t * NE + p] = i;
    }
}

// ---------------- kernel: fused edge MLP + gather-multiply-scatter ----------------
// blockIdx.y = type t, blockIdx.x = 16-edge tile; 128 threads = 4 waves
__global__ void edge_kernel(const float* __restrict__ dist, const int* __restrict__ senders,
                            const int* __restrict__ receivers, const int* __restrict__ counts,
                            const int* __restrict__ elist,
                            const unsigned short* __restrict__ w1t_base,
                            const unsigned short* __restrict__ w2t_base,
                            const float* b1s, const float* b1a, const float* b1n,
                            const float* b2s, const float* b2a, const float* b2n,
                            const float* __restrict__ hx, float* __restrict__ z) {
    int t    = blockIdx.y;
    int cnt  = counts[t];
    int base = blockIdx.x * 16;
    if (base >= cnt) return;

    const unsigned short* w1t = w1t_base + t * DD * DD;
    const unsigned short* w2t = w2t_base + t * KF * DD;
    const float* b1 = (t == 0) ? b1s : ((t == 1) ? b1a : b1n);
    const float* b2 = (t == 0) ? b2s : ((t == 1) ? b2a : b2n);
    float* zt = z + (size_t)t * NNODES * KF;

    constexpr int LDA = DD + 8;
    __shared__ unsigned short a_lds[16 * LDA];   // dist tile, bf16
    __shared__ unsigned short h_lds[16 * LDA];   // ssp(h1) tile, bf16
    __shared__ int eid[16], snd[16], rcv[16], vld[16];

    int tid = threadIdx.x, lane = tid & 31, w = tid >> 5;

    if (tid < 16) {
        int ok = (base + tid) < cnt;
        int e  = elist[t * NE + (ok ? base + tid : base)];
        eid[tid] = e; vld[tid] = ok;
        snd[tid] = senders[e]; rcv[tid] = receivers[e];
    }
    __syncthreads();
    for (int i = tid; i < 16 * DD; i += 128) {
        int r = i >> 6, cidx = i & 63;
        a_lds[r * LDA + cidx] = f2bf(dist[(size_t)eid[r] * DD + cidx]);
    }
    __syncthreads();

    // GEMM1: h1 = ssp(dist @ W1 + b1), each wave one 16x16 tile of N=64
    int n0 = w * 16, nn = n0 + (lane & 15);
    v8f c; float bias1 = b1[nn];
#pragma unroll
    for (int r = 0; r < 8; ++r) c[r] = bias1;
    for (int k0 = 0; k0 < DD; k0 += 32) {
        c = wmma_bf16(load_a_frag(a_lds, LDA, k0, lane),
                      load_b_frag(w1t, DD, n0, k0, lane), c);
    }
    int mh = (lane >> 4) * 8;
#pragma unroll
    for (int r = 0; r < 8; ++r) h_lds[(mh + r) * LDA + nn] = f2bf(ssp(c[r]));
    __syncthreads();

    // GEMM2: we = h1 @ W2 + b2 (16x128), 2 column tiles per wave; fused epilogue:
    // weh = we * hx[sender], atomically accumulated into z[t][receiver]
#pragma unroll
    for (int half = 0; half < 2; ++half) {
        int cn0 = half * 64 + w * 16;
        int cn  = cn0 + (lane & 15);
        v8f c2; float bias2 = b2[cn];
#pragma unroll
        for (int r = 0; r < 8; ++r) c2[r] = bias2;
        for (int k0 = 0; k0 < DD; k0 += 32) {
            c2 = wmma_bf16(load_a_frag(h_lds, LDA, k0, lane),
                           load_b_frag(w2t, DD, cn0, k0, lane), c2);
        }
#pragma unroll
        for (int r = 0; r < 8; ++r) {
            int m = mh + r;
            if (vld[m]) {
                float v = c2[r] * hx[(size_t)snd[m] * KF + cn];
                atomicAdd(&zt[(size_t)rcv[m] * KF + cn], v);
            }
        }
    }
}

// ---------------- kernel: out = elec + sum_t z_t[elec rows] @ gW_t + gb_t ----------------
// one batch per block; 256 threads = 8 waves, each 2 column tiles of N=256
__global__ void out_gemm(const float* __restrict__ elec, const float* __restrict__ z,
                         const unsigned short* __restrict__ gWt_base,
                         const float* gbs, const float* gba, const float* gbn,
                         float* __restrict__ out) {
    constexpr int LDA = KF + 8;
    __shared__ unsigned short a_lds[16 * LDA];
    int b = blockIdx.x, tid = threadIdx.x, lane = tid & 31, w = tid >> 5;

    int n0a = w * 16, n0b = 128 + w * 16;
    int na = n0a + (lane & 15), nb = n0b + (lane & 15);
    v8f c0, c1;
    float ba = gbs[na] + gba[na] + gbn[na];
    float bb = gbs[nb] + gba[nb] + gbn[nb];
#pragma unroll
    for (int r = 0; r < 8; ++r) { c0[r] = ba; c1[r] = bb; }

    for (int t = 0; t < 3; ++t) {
        const float* zt = z + ((size_t)t * NNODES + (size_t)b * NPART + NNUC) * KF;
        const unsigned short* gWt = gWt_base + t * EMBD * KF;
        __syncthreads();
        for (int i = tid; i < 16 * KF; i += 256) {
            int r = i >> 7, cc = i & 127;
            a_lds[r * LDA + cc] = f2bf(zt[i]);
        }
        __syncthreads();
        for (int k0 = 0; k0 < KF; k0 += 32) {
            v16bf a = load_a_frag(a_lds, LDA, k0, lane);
            c0 = wmma_bf16(a, load_b_frag(gWt, KF, n0a, k0, lane), c0);
            c1 = wmma_bf16(a, load_b_frag(gWt, KF, n0b, k0, lane), c1);
        }
    }

    int mh = (lane >> 4) * 8;
    const float* esrc = elec + (size_t)b * NELEC * EMBD;
    float* dst = out + (size_t)b * NELEC * EMBD;
#pragma unroll
    for (int r = 0; r < 8; ++r) {
        int m = mh + r;
        dst[(size_t)m * EMBD + na] = esrc[(size_t)m * EMBD + na] + c0[r];
        dst[(size_t)m * EMBD + nb] = esrc[(size_t)m * EMBD + nb] + c1[r];
    }
}

// ---------------- launch ----------------
extern "C" void kernel_launch(void* const* d_in, const int* in_sizes, int n_in,
                              void* d_out, int out_size, void* d_ws, size_t ws_size,
                              hipStream_t stream) {
    const float* nuc       = (const float*)d_in[0];
    const float* elec      = (const float*)d_in[1];
    const float* dist      = (const float*)d_in[2];
    const int*   e_type    = (const int*)d_in[3];
    const int*   senders   = (const int*)d_in[4];
    const int*   receivers = (const int*)d_in[5];
    // weight order in dict: same, anti, n  -> type index 0,1,2
    const float* wW1[3] = {(const float*)d_in[6],  (const float*)d_in[10], (const float*)d_in[14]};
    const float* wb1[3] = {(const float*)d_in[7],  (const float*)d_in[11], (const float*)d_in[15]};
    const float* wW2[3] = {(const float*)d_in[8],  (const float*)d_in[12], (const float*)d_in[16]};
    const float* wb2[3] = {(const float*)d_in[9],  (const float*)d_in[13], (const float*)d_in[17]};
    const float* hW  = (const float*)d_in[18];
    const float* hb  = (const float*)d_in[19];
    const float* gW[3] = {(const float*)d_in[20], (const float*)d_in[22], (const float*)d_in[24]};
    const float* gb[3] = {(const float*)d_in[21], (const float*)d_in[23], (const float*)d_in[25]};

    // ---- workspace carve-up (~84 MB) ----
    float* hx   = (float*)d_ws;                                // NNODES*KF f32   (20 MB)
    float* z    = hx + (size_t)NNODES * KF;                    // 3*NNODES*KF f32 (60 MB)
    int*   counts = (int*)(z + (size_t)3 * NNODES * KF);       // 16 ints
    int*   elist  = counts + 16;                               // 3*NE ints (3.7 MB)
    unsigned short* w1t = (unsigned short*)(elist + (size_t)3 * NE);
    unsigned short* w2t = w1t + 3 * DD * DD;
    unsigned short* hWt = w2t + 3 * KF * DD;
    unsigned short* gWt = hWt + EMBD * KF;

    // transposed-bf16 weight conversion (tiny, L2-resident afterwards)
    for (int t = 0; t < 3; ++t) {
        cvt_transpose<<<(DD * DD + 255) / 256, 256, 0, stream>>>(wW1[t], w1t + t * DD * DD, DD, DD);
        cvt_transpose<<<(DD * KF + 255) / 256, 256, 0, stream>>>(wW2[t], w2t + t * KF * DD, DD, KF);
        cvt_transpose<<<(KF * EMBD + 255) / 256, 256, 0, stream>>>(gW[t], gWt + t * EMBD * KF, KF, EMBD);
    }
    cvt_transpose<<<(EMBD * KF + 255) / 256, 256, 0, stream>>>(hW, hWt, EMBD, KF);

    hipMemsetAsync((void*)z, 0, (size_t)3 * NNODES * KF * sizeof(float), stream);
    hipMemsetAsync((void*)counts, 0, 16 * sizeof(int), stream);

    copy_nuc<<<(NB * NNUC * KF + 255) / 256, 256, 0, stream>>>(nuc, hx);
    hx_gemm<<<NB, 256, 0, stream>>>(elec, hWt, hb, hx);
    compact_edges<<<(NE + 255) / 256, 256, 0, stream>>>(e_type, counts, elist);

    dim3 eg((NE + 15) / 16, 3);
    edge_kernel<<<eg, 128, 0, stream>>>(dist, senders, receivers, counts, elist,
                                        w1t, w2t,
                                        wb1[0], wb1[1], wb1[2],
                                        wb2[0], wb2[1], wb2[2],
                                        hx, z);

    out_gemm<<<NB, 256, 0, stream>>>(elec, z, gWt, gb[0], gb[1], gb[2], (float*)d_out);
}
